// SAGE_27676769256198
// MI455X (gfx1250) — matrix operand order repaired and checked by reading, not compiled, not run on hardware
//
#include <hip/hip_runtime.h>
#include <hip/hip_bf16.h>

// ---------------------------------------------------------------------------
// Heterogeneous GraphSAGE for MI455X (gfx1250, wave32).
// Strategy: aggregate-then-GEMM. Per relation:
//   agg[dst] += h_src[src] (edge atomics), deg[dst] += 1
//   out     += [h_dst | agg/deg] @ [W_self ; W_neigh] + b     (bf16 WMMA, f32 acc)
// Word target sums 3 relations, doc target 1, then ReLU; 2 layers; sigmoid head.
// LDS tiles are stored in WMMA *fragment order* so each lane fetches its whole
// v16bf fragment with contiguous 32-byte LDS loads (ds_load_b128 pairs).
// ---------------------------------------------------------------------------

typedef __attribute__((ext_vector_type(16))) __bf16 v16bf;
typedef __attribute__((ext_vector_type(4)))  __bf16 v4bf;
typedef __attribute__((ext_vector_type(8)))  float  v8f;

#define NW 50000
#define ND 10000
#define DH 128

// ---------------- utility kernels ----------------
__global__ void zero_kernel(float* __restrict__ p, size_t n) {
    size_t i = (size_t)blockIdx.x * blockDim.x + threadIdx.x;
    size_t stride = (size_t)gridDim.x * blockDim.x;
    for (; i < n; i += stride) p[i] = 0.0f;
}

__global__ void relu_kernel(float* __restrict__ p, size_t n) {
    size_t i = (size_t)blockIdx.x * blockDim.x + threadIdx.x;
    size_t stride = (size_t)gridDim.x * blockDim.x;
    for (; i < n; i += stride) p[i] = fmaxf(p[i], 0.0f);
}

// ---------------- edge aggregation ----------------
// one wave32 per edge; lanes cover the feature row in float4 strips
__global__ __launch_bounds__(256) void aggregate_kernel(
    const float* __restrict__ hsrc, const int* __restrict__ src,
    const int* __restrict__ dst, float* __restrict__ agg,
    float* __restrict__ deg, int E, int d)
{
    int e = blockIdx.x * 8 + (threadIdx.x >> 5);
    int lane = threadIdx.x & 31;
    if (e >= E) return;
    int s = src[e];
    int t = dst[e];
    if (lane == 0) atomicAdd(deg + t, 1.0f);
    const float* sp = hsrc + (size_t)s * d;
    float*       tp = agg  + (size_t)t * d;
    for (int c0 = lane * 4; c0 < d; c0 += 128) {
        float4 v = *(const float4*)(sp + c0);
        atomicAdd(tp + c0 + 0, v.x);
        atomicAdd(tp + c0 + 1, v.y);
        atomicAdd(tp + c0 + 2, v.z);
        atomicAdd(tp + c0 + 3, v.w);
    }
}

// ---------------- fused SAGE GEMM (bf16 WMMA, f32 accumulate) ----------------
// out[n_dst,128] += hself @ Wself + (agg/max(deg,1)) @ Wneigh + bias
// Block tile: 32 rows x 128 cols. 8 waves; wave w: m-tile = w>>2 (16 rows),
// n-tiles = (w&3)*2, (w&3)*2+1. K = 2*d, stepped by 32.
//
// LDS layout = WMMA fragment order:
//   A (16-bit 16x32, contiguous k, not ISA table order): elem (m,k) -> lane (m&15)+16*(k>>4),
//                                 slot e = k&15
//   B (16-bit 32x16):             elem (k,n) -> tile n>>4,
//                                 lane (n&15)+16*(k>>4), slot e = k&15
__global__ __launch_bounds__(256) void sage_gemm_kernel(
    const float* __restrict__ hself, const float* __restrict__ agg,
    const float* __restrict__ deg,   const float* __restrict__ Wself,
    const float* __restrict__ Wneigh,const float* __restrict__ bias,
    float* __restrict__ out, int n_dst, int d)
{
    __shared__ alignas(32) __bf16 Afrag[2][32][16];   // [m-tile][lane][slot]  2 KB
    __shared__ alignas(32) __bf16 Bfrag[8][32][16];   // [n-tile][lane][slot]  8 KB

    const int tid  = threadIdx.x;
    const int wid  = tid >> 5;
    const int lane = tid & 31;
    const int row0 = blockIdx.x * 32;

    const int mt  = wid >> 2;        // wave's m-tile: 0 or 1
    const int nt0 = (wid & 3) * 2;   // wave's first n-tile: 0,2,4,6

    v8f c0 = {};
    v8f c1 = {};

    // ---- A loader mapping: thread -> (row, 4 consecutive k) ----
    const int arow   = tid >> 3;           // 0..31
    const int acol0  = (tid & 7) * 4;      // 0,4,...,28 (same k-half for all 4)
    const int a_mt   = arow >> 4;
    const int a_lane = (arow & 15) + ((acol0 >> 4) << 4);
    const int a_e    = acol0 & 15;
    const int grow   = row0 + arow;

    // ---- B loader mapping: thread -> one (n-tile, lane) fragment row ----
    const int bnt   = wid;                     // 0..7
    const int blane = lane;                    // 0..31
    const int bn    = bnt * 16 + (blane & 15); // global col 0..127
    const int bkoff = (blane >> 4) * 16;       // 0 or 16 within K-step

    // reciprocal degree once per row (v_rcp_f32, exact enough: deg is integral)
    float rdeg = 0.0f;
    if (grow < n_dst) rdeg = __builtin_amdgcn_rcpf(fmaxf(deg[grow], 1.0f));

    const int K = 2 * d;
    for (int k0 = 0; k0 < K; k0 += 32) {
        // ---- stage A tile (fused [h_self | mean]) into fragment-order LDS ----
        {
            float v0 = 0.f, v1 = 0.f, v2 = 0.f, v3 = 0.f;
            if (grow < n_dst) {
                int kk0 = k0 + acol0;          // d % 32 == 0 -> no straddle
                if (kk0 < d) {
                    const float* p = hself + (size_t)grow * d + kk0;
                    v0 = p[0]; v1 = p[1]; v2 = p[2]; v3 = p[3];
                } else {
                    const float* p = agg + (size_t)grow * d + (kk0 - d);
                    v0 = p[0] * rdeg; v1 = p[1] * rdeg;
                    v2 = p[2] * rdeg; v3 = p[3] * rdeg;
                }
            }
            v4bf pk = { (__bf16)v0, (__bf16)v1, (__bf16)v2, (__bf16)v3 };
            *(v4bf*)&Afrag[a_mt][a_lane][a_e] = pk;   // one 8B ds_store
        }
        // ---- stage B tile into fragment-order LDS (one 32B store/thread) ----
        {
            int kkb = k0 + bkoff;
            const float* base = (kkb < d)
                ? (Wself  + (size_t)kkb       * DH + bn)
                : (Wneigh + (size_t)(kkb - d) * DH + bn);
            v16bf pk;
            #pragma unroll
            for (int e = 0; e < 16; ++e) pk[e] = (__bf16)base[(size_t)e * DH];
            *(v16bf*)&Bfrag[bnt][blane][0] = pk;
            // prefetch next K-step's weight column (global_prefetch_b8)
            if (kkb + 32 < K) {
                const float* np = ((kkb + 32) < d)
                    ? (Wself  + (size_t)(kkb + 32)     * DH + bn)
                    : (Wneigh + (size_t)(kkb + 32 - d) * DH + bn);
                __builtin_prefetch(np, 0, 1);
            }
        }
        __syncthreads();

        // ---- fragment loads: contiguous 32B per lane (ds_load_b128 x2) ----
        v16bf af  = *(const v16bf*)&Afrag[mt][lane][0];
        v16bf bf0 = *(const v16bf*)&Bfrag[nt0][lane][0];
        v16bf bf1 = *(const v16bf*)&Bfrag[nt0 + 1][lane][0];

        c0 = __builtin_amdgcn_wmma_f32_16x16x32_bf16(
                 false, af, false, bf0, (short)0, c0, false, false);
        c1 = __builtin_amdgcn_wmma_f32_16x16x32_bf16(
                 false, af, false, bf1, (short)0, c1, false, false);
        __syncthreads();
    }

    // ---- epilogue: accumulate into out + per-relation bias ----
    // C/D layout: lanes 0-15 N=lane, lanes 16-31 N=lane-16 / M += 8; VGPR r -> M=r
    {
        int n0    = nt0 * 16 + (lane & 15);
        int n1    = n0 + 16;
        int rbase = mt * 16 + (lane >> 4) * 8;
        float b0 = bias[n0];
        float b1 = bias[n1];
        #pragma unroll
        for (int r = 0; r < 8; ++r) {
            int growc = row0 + rbase + r;
            if (growc < n_dst) {
                out[(size_t)growc * DH + n0] += c0[r] + b0;
                out[(size_t)growc * DH + n1] += c1[r] + b1;
            }
        }
    }
}

// ---------------- sigmoid head ----------------
__global__ void head_kernel(const float* __restrict__ x,
                            const float* __restrict__ w,
                            const float* __restrict__ b,
                            float* __restrict__ out, int n)
{
    int i = blockIdx.x * blockDim.x + threadIdx.x;
    if (i >= n) return;
    const float4* xr = (const float4*)(x + (size_t)i * DH);
    const float4* wr = (const float4*)w;
    float acc = 0.0f;
    #pragma unroll
    for (int j = 0; j < DH / 4; ++j) {
        float4 a = xr[j];
        float4 c = wr[j];
        acc += a.x * c.x + a.y * c.y + a.z * c.z + a.w * c.w;
    }
    acc += b[0];
    out[i] = 1.0f / (1.0f + __expf(-acc));
}

// ---------------------------------------------------------------------------
extern "C" void kernel_launch(void* const* d_in, const int* in_sizes, int n_in,
                              void* d_out, int out_size, void* d_ws, size_t ws_size,
                              hipStream_t stream) {
    const float* x_word = (const float*)d_in[0];
    const float* x_doc  = (const float*)d_in[1];
    const int* ww_src  = (const int*)d_in[2];
    const int* ww_dst  = (const int*)d_in[3];
    const int* wwr_src = (const int*)d_in[4];
    const int* wwr_dst = (const int*)d_in[5];
    const int* wd_src  = (const int*)d_in[6];
    const int* wd_dst  = (const int*)d_in[7];
    const int* wdr_src = (const int*)d_in[8];
    const int* wdr_dst = (const int*)d_in[9];
    const float* Ws1 = (const float*)d_in[10];  // [4,256,128]
    const float* Wn1 = (const float*)d_in[11];
    const float* b1  = (const float*)d_in[12];  // [4,128]
    const float* Ws2 = (const float*)d_in[13];  // [4,128,128]
    const float* Wn2 = (const float*)d_in[14];
    const float* b2  = (const float*)d_in[15];
    const float* lin_w = (const float*)d_in[16];
    const float* lin_b = (const float*)d_in[17];

    const int E_ww  = in_sizes[2];
    const int E_wwr = in_sizes[4];
    const int E_wd  = in_sizes[6];
    const int E_wdr = in_sizes[8];

    // workspace carve-up (floats)
    float* ws = (float*)d_ws;
    size_t off = 0;
    float* agg = ws + off; off += (size_t)NW * 256;   // max agg (word target, d=256)
    float* deg = ws + off; off += NW;
    float* xw1 = ws + off; off += (size_t)NW * DH;
    float* xd1 = ws + off; off += (size_t)ND * DH;
    float* xw2 = ws + off; off += (size_t)NW * DH;
    float* xd2 = ws + off; off += (size_t)ND * DH;
    (void)ws_size; (void)n_in;

    const size_t W1 = (size_t)256 * DH;  // per-relation weight stride, layer 1
    const size_t W2 = (size_t)DH  * DH;  // layer 2

    auto run_rel = [&](const float* hsrc, const float* hdst,
                       const int* src, const int* dst, int E,
                       int n_dst, int d,
                       const float* Wself, const float* Wneigh,
                       const float* bias, float* outp) {
        zero_kernel<<<1024, 256, 0, stream>>>(agg, (size_t)n_dst * d);
        zero_kernel<<<256, 256, 0, stream>>>(deg, (size_t)n_dst);
        aggregate_kernel<<<(E + 7) / 8, 256, 0, stream>>>(hsrc, src, dst, agg, deg, E, d);
        sage_gemm_kernel<<<(n_dst + 31) / 32, 256, 0, stream>>>(
            hdst, agg, deg, Wself, Wneigh, bias, outp, n_dst, d);
    };

    // ===================== layer 1 (d_in = 256) =====================
    // word target: ww (rel 0) + wwr (rel 1) + wdr (rel 3)
    zero_kernel<<<1024, 256, 0, stream>>>(xw1, (size_t)NW * DH);
    run_rel(x_word, x_word, ww_src,  ww_dst,  E_ww,  NW, 256,
            Ws1 + 0 * W1, Wn1 + 0 * W1, b1 + 0 * DH, xw1);
    run_rel(x_word, x_word, wwr_src, wwr_dst, E_wwr, NW, 256,
            Ws1 + 1 * W1, Wn1 + 1 * W1, b1 + 1 * DH, xw1);
    run_rel(x_doc,  x_word, wdr_src, wdr_dst, E_wdr, NW, 256,
            Ws1 + 3 * W1, Wn1 + 3 * W1, b1 + 3 * DH, xw1);
    relu_kernel<<<1024, 256, 0, stream>>>(xw1, (size_t)NW * DH);
    // doc target: wd (rel 2)
    zero_kernel<<<1024, 256, 0, stream>>>(xd1, (size_t)ND * DH);
    run_rel(x_word, x_doc, wd_src, wd_dst, E_wd, ND, 256,
            Ws1 + 2 * W1, Wn1 + 2 * W1, b1 + 2 * DH, xd1);
    relu_kernel<<<1024, 256, 0, stream>>>(xd1, (size_t)ND * DH);

    // ===================== layer 2 (d = 128) =====================
    zero_kernel<<<1024, 256, 0, stream>>>(xw2, (size_t)NW * DH);
    run_rel(xw1, xw1, ww_src,  ww_dst,  E_ww,  NW, DH,
            Ws2 + 0 * W2, Wn2 + 0 * W2, b2 + 0 * DH, xw2);
    run_rel(xw1, xw1, wwr_src, wwr_dst, E_wwr, NW, DH,
            Ws2 + 1 * W2, Wn2 + 1 * W2, b2 + 1 * DH, xw2);
    run_rel(xd1, xw1, wdr_src, wdr_dst, E_wdr, NW, DH,
            Ws2 + 3 * W2, Wn2 + 3 * W2, b2 + 3 * DH, xw2);
    relu_kernel<<<1024, 256, 0, stream>>>(xw2, (size_t)NW * DH);
    zero_kernel<<<1024, 256, 0, stream>>>(xd2, (size_t)ND * DH);
    run_rel(xw1, xd1, wd_src, wd_dst, E_wd, ND, DH,
            Ws2 + 2 * W2, Wn2 + 2 * W2, b2 + 2 * DH, xd2);
    relu_kernel<<<1024, 256, 0, stream>>>(xd2, (size_t)ND * DH);

    // ===================== head =====================
    float* out = (float*)d_out;
    head_kernel<<<(NW + 255) / 256, 256, 0, stream>>>(xw2, lin_w, lin_b, out, NW);
    head_kernel<<<(ND + 255) / 256, 256, 0, stream>>>(xd2, lin_w, lin_b, out + NW, ND);
    (void)out_size;
}
